// LSTMModel_84112639525712
// MI455X (gfx1250) — matrix operand order repaired and checked by reading, not compiled, not run on hardware
//
#include <hip/hip_runtime.h>

typedef __attribute__((ext_vector_type(16))) __bf16 v16bf;
typedef __attribute__((ext_vector_type(8)))  __bf16 v8bf;
typedef __attribute__((ext_vector_type(8)))  float  v8f;
typedef __attribute__((ext_vector_type(2)))  float  v2f;

#define LSTM_B   256
#define LSTM_C   2000
#define LSTM_N   (LSTM_B * LSTM_C)   // 512000 independent sequences
#define LSTM_S   50
#define LSTM_H   64
#define LSTM_G   (4 * LSTM_H)        // 256 gate columns (i,f,g,o)
#define LSTM_NC  3

#define WAVES            8
#define ROWS_PER_WAVE    16
#define ROWS_PER_BLOCK   (WAVES * ROWS_PER_WAVE)  // 128
#define XPAD             52                        // padded x stride (bank spread)

__device__ __forceinline__ float fast_sigmoid(float x) {
    // 1 / (1 + 2^(-x*log2 e)) : v_exp_f32 + v_rcp_f32 on the TRANS pipe
    return __builtin_amdgcn_rcpf(1.0f + __builtin_amdgcn_exp2f(-1.4426950408889634f * x));
}
__device__ __forceinline__ float fast_tanh(float x) {
    // tanh(x) = 2*sigmoid(2x) - 1
    return 2.0f * __builtin_amdgcn_rcpf(1.0f + __builtin_amdgcn_exp2f(-2.8853900817779268f * x)) - 1.0f;
}

__global__ __launch_bounds__(ROWS_PER_BLOCK * 2)  // 256 threads = 8 waves
void lstm_wmma_bf16_kernel(const float* __restrict__ x,
                           const float* __restrict__ W_ih,
                           const float* __restrict__ W_hh,
                           const float* __restrict__ b_ih,
                           const float* __restrict__ b_hh,
                           const float* __restrict__ W_fc,
                           const float* __restrict__ b_fc,
                           float* __restrict__ out)
{
    // W_hh as bf16, row-major [g][k]: B-fragment reads are two 16B runs per lane.
    __shared__ __bf16 whlds[LSTM_G * LSTM_H];                       // 32 KB
    // Per-wave private h staging (bf16) used to transpose D-layout -> A-layout.
    __shared__ __bf16 hlds[WAVES][ROWS_PER_WAVE * LSTM_H];          // 16 KB
    // Per-wave x tile [16 rows][50 steps].
    __shared__ float  xlds[WAVES][ROWS_PER_WAVE * XPAD];            // ~26.6 KB

    const int tid  = threadIdx.x;
    const int wave = tid >> 5;
    const int lane = tid & 31;
    const int nn   = lane & 15;        // column-within-tile / A-matrix row
    const int hg   = (lane >> 4) & 1;  // half-wave group

    // ---- Stage W_hh (f32 -> bf16) cooperatively ----
    for (int i = tid; i < LSTM_G * LSTM_H; i += blockDim.x)
        whlds[i] = (__bf16)W_hh[i];

    const int seq0 = blockIdx.x * ROWS_PER_BLOCK + wave * ROWS_PER_WAVE;

    // ---- Stage this wave's x tile ----
    for (int i = lane; i < ROWS_PER_WAVE * LSTM_S; i += 32) {
        int r = i / LSTM_S, t = i % LSTM_S;
        xlds[wave][r * XPAD + t] = x[(size_t)(seq0 + r) * LSTM_S + t];
    }
    // ---- Zero h staging (wave-private; DS ops are in-order per wave) ----
    for (int i = lane; i < ROWS_PER_WAVE * LSTM_H; i += 32)
        hlds[wave][i] = (__bf16)0.0f;

    __syncthreads();  // make whlds visible to all waves

    // B3 fragments for the f32 16x16x4 WMMA that folds x*W_ih + bias into the
    // matrix pipe at full f32 precision. B(4x16, KxN): VGPR0 holds the K=0 row
    // for lanes 0-15 (K=2 for lanes 16-31), VGPR1 the K=1 row (K=3 for 16-31).
    // Rows K>=2 are zero, so hg=1 lanes carry zeros.
    v2f b3[16];
    #pragma unroll
    for (int ct = 0; ct < 16; ++ct) {
        int g = ct * 16 + nn;
        float w = W_ih[g];
        float b = b_ih[g] + b_hh[g];
        b3[ct][0] = hg ? 0.0f : w;   // K=0 row: W_ih column
        b3[ct][1] = hg ? 0.0f : b;   // K=1 row: combined bias (ones column of A)
    }
    const float a3_ones = hg ? 0.0f : 1.0f;   // A(16x4) K=1 column = 1.0

    // c state in D-layout: element r of hidden tile j -> row (r + 8*hg), col 16*j + nn.
    float cst[4][8];
    #pragma unroll
    for (int j = 0; j < 4; ++j)
        #pragma unroll
        for (int r = 0; r < 8; ++r)
            cst[j][r] = 0.0f;

    const __bf16* wl = whlds;
    __bf16*       hl = hlds[wave];
    const float*  xl = xlds[wave];

    for (int t = 0; t < LSTM_S; ++t) {
        // A3 (16x4 f32): K=0 column = x_t for row nn; K=1 column = ones.
        v2f a3;
        float xnn = xl[nn * XPAD + t];
        a3[0] = hg ? 0.0f : xnn;
        a3[1] = a3_ones;

        // A fragments (16x32 bf16, ISA layout): row M = nn;
        // contiguous runs at K = 32kc+8hg and K = 32kc+16+8hg.
        v16bf afrag[2];
        #pragma unroll
        for (int kc = 0; kc < 2; ++kc) {
            v16bf av;
            const __bf16* ap = hl + nn * LSTM_H + 32 * kc + 8 * hg;
            ((v8bf*)&av)[0] = *(const v8bf*)(ap);        // K = 32kc+8hg .. +7
            ((v8bf*)&av)[1] = *(const v8bf*)(ap + 16);   // K = 32kc+16+8hg .. +7
            afrag[kc] = av;
        }

        #pragma unroll
        for (int j = 0; j < 4; ++j) {            // hidden 16-wide tile
            v8f acc[4];                          // i, f, g, o accumulators
            #pragma unroll
            for (int gi = 0; gi < 4; ++gi) {
                const int ct = gi * 4 + j;       // gate block gi, hidden tile j
                // Rank-2 f32 WMMA: gates_init = [x_t | 1] x [W_ih ; bias]
                v8f a = {};
                a = __builtin_amdgcn_wmma_f32_16x16x4_f32(
                        false, a3, false, b3[ct], (short)0, a, false, false);
                #pragma unroll
                for (int kc = 0; kc < 2; ++kc) {
                    // B fragment 32x16 bf16: lane holds K = 32kc+8hg+(0..7) and 32kc+16+8hg+(0..7),
                    // N = 16*ct + nn; two 16B runs in row-major W_hh.
                    v16bf bv;
                    const __bf16* bp = wl + (ct * 16 + nn) * LSTM_H + 32 * kc + 8 * hg;
                    ((v8bf*)&bv)[0] = *(const v8bf*)(bp);
                    ((v8bf*)&bv)[1] = *(const v8bf*)(bp + 16);
                    a = __builtin_amdgcn_wmma_f32_16x16x32_bf16(
                            false, afrag[kc], false, bv, (short)0, a, false, false);
                }
                acc[gi] = a;
            }
            // Elementwise LSTM cell update for hidden tile j.
            #pragma unroll
            for (int r = 0; r < 8; ++r) {
                float ig = fast_sigmoid(acc[0][r]);
                float fg = fast_sigmoid(acc[1][r]);
                float gg = fast_tanh  (acc[2][r]);
                float og = fast_sigmoid(acc[3][r]);
                float c  = fg * cst[j][r] + ig * gg;
                cst[j][r] = c;
                float h  = og * fast_tanh(c);
                hl[(r + 8 * hg) * LSTM_H + j * 16 + nn] = (__bf16)h;
            }
        }
    }

    // ---- Final FC: logits[16 rows][3] = h_last @ W_fc.T + b_fc ----
    #pragma unroll
    for (int base = 0; base < 64; base += 32) {
        int idx = base + lane;
        if (idx < ROWS_PER_WAVE * LSTM_NC) {
            int r = idx / LSTM_NC, cls = idx % LSTM_NC;
            float s = b_fc[cls];
            #pragma unroll 8
            for (int k = 0; k < LSTM_H; ++k)
                s += (float)hl[r * LSTM_H + k] * W_fc[cls * LSTM_H + k];
            out[(size_t)(seq0 + r) * LSTM_NC + cls] = s;
        }
    }
}

extern "C" void kernel_launch(void* const* d_in, const int* in_sizes, int n_in,
                              void* d_out, int out_size, void* d_ws, size_t ws_size,
                              hipStream_t stream) {
    (void)in_sizes; (void)n_in; (void)out_size; (void)d_ws; (void)ws_size;
    const float* x    = (const float*)d_in[0];
    const float* W_ih = (const float*)d_in[1];
    const float* W_hh = (const float*)d_in[2];
    const float* b_ih = (const float*)d_in[3];
    const float* b_hh = (const float*)d_in[4];
    const float* W_fc = (const float*)d_in[5];
    const float* b_fc = (const float*)d_in[6];
    float* out = (float*)d_out;

    dim3 grid(LSTM_N / ROWS_PER_BLOCK);   // 4000 blocks
    dim3 block(ROWS_PER_BLOCK * 2);       // 256 threads = 8 wave32
    lstm_wmma_bf16_kernel<<<grid, block, 0, stream>>>(
        x, W_ih, W_hh, b_ih, b_hh, W_fc, b_fc, out);
}
